// dMag0Grid_41205916238514
// MI455X (gfx1250) — compile-verified, matrix operand order checked by reading, not run on hardware
//
#include <hip/hip_runtime.h>

#define N_ORB    1024
#define NTIMES   1024
#define N_FZ     24
#define N_KEZ    6
#define N_TINT   64
#define N_ALPHA  512

#define TINT_TILE   16          // tints per block
#define TILE_STRIDE 516         // 512 + 2x2 TDM pad -> 16 distinct LDS banks across M
#define THREADS     256         // 8 wave32s
#define WAVES       8
#define CHUNKS_PER_WAVE 2       // 16 chunks of 64 orbits / 8 waves

typedef __attribute__((ext_vector_type(8))) int          v8i;
typedef __attribute__((ext_vector_type(4))) unsigned int u32x4;
typedef __attribute__((ext_vector_type(8))) int          i32x8;
typedef __attribute__((ext_vector_type(4))) int          i32x4;

// Padded in-row index: TDM inserts 2 DWORDs after every 256 DWORDs stored.
__device__ __forceinline__ int pad_idx(int s) { return s + ((s >> 8) << 1); }

__global__ __launch_bounds__(THREADS) void pdet_kernel(
    const float* __restrict__ alpha,    // (N_ORB, NTIMES)
    const float* __restrict__ dMag,     // (N_ORB, NTIMES)
    const float* __restrict__ fZ_vals,  // (NTIMES,)
    const float* __restrict__ kEZ_val_p,// (1,)
    const float* __restrict__ grid,     // (N_FZ, N_KEZ, N_TINT, N_ALPHA)
    const float* __restrict__ fZs,      // (N_FZ,)
    const float* __restrict__ kEZs,     // (N_KEZ,)
    const float* __restrict__ alphas,   // (N_ALPHA,)
    float* __restrict__ out)            // (NTIMES, N_TINT)
{
    __shared__ float tile[TINT_TILE * TILE_STRIDE]; // 33 KB grid tile (TDM-padded)
    __shared__ int   sArr[N_ORB];                   // 4 KB
    __shared__ float daArr[N_ORB];                  // 4 KB
    __shared__ float dmArr[N_ORB];                  // 4 KB (dMag, +inf when !geom)
    __shared__ int   partial[WAVES * TINT_TILE];    // cross-wave count reduction

    const int tid      = threadIdx.x;
    const int blk      = blockIdx.x;        // NTIMES * 4 blocks
    const int t        = blk >> 2;          // time index
    const int tintBase = (blk & 3) * TINT_TILE;
    const int wave     = tid >> 5;
    const int lane     = tid & 31;

    // ---- uniform scalars (cheap; computed by every thread) ----
    const float log_fZ0  = log10f(fZs[0]);
    const float inv_dlfZ = 1.0f / (log10f(fZs[1]) - log_fZ0);
    const float log_al0  = log10f(alphas[0]);
    const float inv_dlal = 1.0f / (log10f(alphas[1]) - log_al0);
    const float alpha_lo = alphas[0];
    const float alpha_hi = alphas[N_ALPHA - 1];
    const float kEZ_val  = kEZ_val_p[0];

    // searchsorted(kEZs, kEZ_val, side='right') - 1, clipped (jnp.take mode='clip')
    int kEZ_ind = -1;
    #pragma unroll
    for (int i = 0; i < N_KEZ; ++i) kEZ_ind += (kEZs[i] <= kEZ_val) ? 1 : 0;
    kEZ_ind = kEZ_ind < 0 ? 0 : (kEZ_ind > N_KEZ - 1 ? N_KEZ - 1 : kEZ_ind);

    // fZ0 = clip(floor(fZ_ind) + 1, 0, N_FZ-2)
    const float fZ_ind = (log10f(fZ_vals[t]) - log_fZ0) * inv_dlfZ;
    int fz0 = (int)floorf(fZ_ind) + 1;
    fz0 = fz0 < 0 ? 0 : (fz0 > N_FZ - 2 ? N_FZ - 2 : fz0);

    const float* gsrc =
        grid + (((size_t)fz0 * N_KEZ + kEZ_ind) * N_TINT + tintBase) * N_ALPHA;

    // ---- Phase 1: TDM async 2D tile load (16 x 512 f32) into padded LDS ----
#if __has_builtin(__builtin_amdgcn_tensor_load_to_lds)
    if (wave == 0) {
        const unsigned           lds_addr = (unsigned)(unsigned long long)&tile[0];
        const unsigned long long ga       = (unsigned long long)gsrc;
        // D# group0: count=1 | lds_addr | global_addr[56:0] | type=2
        const u32x4 g0 = { 1u,
                           lds_addr,
                           (unsigned)(ga & 0xFFFFFFFFull),
                           (unsigned)((ga >> 32) & 0x01FFFFFFull) | 0x80000000u };
        // D# group1: data_size=4B(2<<16) | pad_en(1<<20) | pad_interval=7(256dw)
        //            | pad_amount=1(2dw); dims: tensor 512x16, tile 512x16, stride0=512
        const i32x8 g1 = { (int)0x03D20000,
                           (int)(512u << 16),   // tensor_dim0[15:0] @ bits 63:48
                           (int)(16u  << 16),   // tensor_dim1[15:0] @ bits 95:80
                           (int)(512u << 16),   // tile_dim0        @ bits 127:112
                           16,                  // tile_dim1        @ bits 143:128
                           512,                 // tensor_dim0_stride[31:0]
                           0, 0 };
        const i32x4 gz = { 0, 0, 0, 0 };
#if defined(__clang_major__) && (__clang_major__ >= 23)
        const i32x8 gz8 = { 0, 0, 0, 0, 0, 0, 0, 0 };
        __builtin_amdgcn_tensor_load_to_lds(g0, g1, gz, gz, gz8, 0);
#else
        __builtin_amdgcn_tensor_load_to_lds(g0, g1, gz, gz, 0);
#endif
    }
#else
    // Fallback: manual staged copy into the same padded layout.
    for (int i = tid; i < TINT_TILE * N_ALPHA; i += THREADS) {
        const int row = i >> 9, col = i & (N_ALPHA - 1);
        tile[row * TILE_STRIDE + pad_idx(col)] = gsrc[i];
    }
#endif

    // ---- Phase 2 (overlaps the TDM DMA): per-orbit index/weight/threshold ----
    #pragma unroll
    for (int it = 0; it < N_ORB / THREADS; ++it) {
        const int orb = it * THREADS + tid;
        const float a   = alpha[(size_t)orb * NTIMES + t];
        const float dm  = dMag [(size_t)orb * NTIMES + t];
        const bool geom = (a >= alpha_lo) && (a <= alpha_hi);
        const float a_ind = (log10f(a) - log_al0) * inv_dlal;
        int a0 = (int)a_ind;                       // trunc toward zero == astype(int32)
        a0 = a0 < 0 ? 0 : (a0 > N_ALPHA - 1 ? N_ALPHA - 1 : a0);
        sArr [orb] = a0 < N_ALPHA - 2 ? a0 : N_ALPHA - 2;   // dynamic_slice clamp
        daArr[orb] = a_ind - (float)a0;
        dmArr[orb] = geom ? dm : __builtin_inff(); // +inf => mask bit always 0
    }

#if __has_builtin(__builtin_amdgcn_tensor_load_to_lds)
    if (wave == 0) __builtin_amdgcn_s_wait_tensorcnt(0);   // DMA done (issuing wave)
#endif
    __syncthreads();                                       // publish tile + orbit arrays

    // ---- Phase 3: build 0/1 A-matrix (16 tints x 64 orbits) and WMMA-reduce ----
    const int M = lane & 15;     // tint row of the A matrix
    const int h = lane >> 4;     // lane half selects K sub-block
    const float* __restrict__ trow = &tile[M * TILE_STRIDE];

    const v8i ones = {0x01010101, 0x01010101, 0x01010101, 0x01010101,
                      0x01010101, 0x01010101, 0x01010101, 0x01010101};
    v8i acc = {};

    #pragma unroll
    for (int cc = 0; cc < CHUNKS_PER_WAVE; ++cc) {
        const int orbBase = (wave * CHUNKS_PER_WAVE + cc) * 64;
        v8i amat;
        #pragma unroll
        for (int v = 0; v < 8; ++v) {
            unsigned pk = 0u;
            #pragma unroll
            for (int b = 0; b < 4; ++b) {
                // 8-bit A 16x64 layout: K = (v/2)*16 + h*8 + (v%2)*4 + b
                const int K   = ((v >> 1) << 4) + (h << 3) + ((v & 1) << 2) + b;
                const int orb = orbBase + K;
                const int   s  = sArr[orb];
                const float da = daArr[orb];
                const float dm = dmArr[orb];
                const float g0 = trow[pad_idx(s)];
                const float g1 = trow[pad_idx(s + 1)];
                const float dim = g0 + da * (g1 - g0);
                const unsigned bit = __float_as_uint(dm - dim) >> 31;  // signbit
                pk |= bit << (8 * b);
            }
            amat[v] = (int)pk;
        }
        // D[M][N] += sum_K A[M][K] * 1  (all columns equal the orbit count)
        acc = __builtin_amdgcn_wmma_i32_16x16x64_iu8(
                  false, amat, false, ones, acc, false, false);
    }

    // Column N=0 of C/D: lane 0 holds rows 0..7 (VGPR r), lane 16 rows 8..15
    if ((lane & 15) == 0) {
        #pragma unroll
        for (int r = 0; r < 8; ++r)
            partial[wave * TINT_TILE + (h ? 8 + r : r)] = acc[r];
    }
    __syncthreads();

    if (tid < TINT_TILE) {
        int total = 0;
        #pragma unroll
        for (int w = 0; w < WAVES; ++w) total += partial[w * TINT_TILE + tid];
        out[(size_t)t * N_TINT + tintBase + tid] = (float)total * (1.0f / (float)N_ORB);
    }
}

extern "C" void kernel_launch(void* const* d_in, const int* in_sizes, int n_in,
                              void* d_out, int out_size, void* d_ws, size_t ws_size,
                              hipStream_t stream) {
    (void)in_sizes; (void)n_in; (void)out_size; (void)d_ws; (void)ws_size;
    const float* alpha   = (const float*)d_in[0];
    const float* dMag    = (const float*)d_in[1];
    const float* fZ_vals = (const float*)d_in[2];
    const float* kEZ_val = (const float*)d_in[3];
    const float* grid    = (const float*)d_in[4];
    const float* fZs     = (const float*)d_in[5];
    const float* kEZs    = (const float*)d_in[6];
    const float* alphas  = (const float*)d_in[7];
    float* out = (float*)d_out;

    dim3 g(NTIMES * (N_TINT / TINT_TILE));   // 4096 blocks: (time, 16-tint group)
    pdet_kernel<<<g, THREADS, 0, stream>>>(alpha, dMag, fZ_vals, kEZ_val,
                                           grid, fZs, kEZs, alphas, out);
}
